// CausalSelfAttention_21672404976176
// MI455X (gfx1250) — matrix operand order repaired and checked by reading, not compiled, not run on hardware
//
#include <hip/hip_runtime.h>
#include <hip/hip_bf16.h>
#include <stdint.h>

typedef unsigned short u16;
typedef __attribute__((ext_vector_type(16))) __bf16 v16bf;
typedef __attribute__((ext_vector_type(8)))  float  v8f;

static constexpr int Tn  = 2048;   // sequence length
static constexpr int Dn  = 1024;   // model dim
static constexpr int Hn  = 16;     // heads
static constexpr int HDn = 64;     // head dim
static constexpr int Bn  = 2;      // batch

#define NEGBIG (-3.0e38f)

union ABfrag {
    uint4 q[2];
    v16bf v;
};

__device__ __forceinline__ v8f wmma_bf16(v16bf a, v16bf b, v8f c) {
    return __builtin_amdgcn_wmma_f32_16x16x32_bf16(false, a, false, b, (short)0, c,
                                                   false, false);
}

__device__ __forceinline__ u16 f2b(float f) {
    uint32_t u = __builtin_bit_cast(uint32_t, f);
    uint32_t r = u + 0x7FFFu + ((u >> 16) & 1u);
    return (u16)(r >> 16);
}

// CDNA5 async memory->LDS copy (ASYNCcnt-tracked, no VGPR round trip).
// lds_off: wave-relative LDS byte address = low 32 bits of the generic pointer
// (ISA 10.2).
__device__ __forceinline__ void async_load_b128(uint32_t lds_off,
                                                const void* gptr) {
    asm volatile("global_load_async_to_lds_b128 %0, %1, off"
                 :
                 : "v"(lds_off), "v"(gptr)
                 : "memory");
}
__device__ __forceinline__ void wait_asynccnt0() {
    asm volatile("s_wait_asynccnt 0x0" ::: "memory");
}

// ---------------------------------------------------------------- fp32 -> bf16
__global__ __launch_bounds__(256) void cvt_f32_bf16(const float* __restrict__ s,
                                                    u16* __restrict__ d, int n) {
    int i = blockIdx.x * 256 + threadIdx.x;
    if (i < n) d[i] = f2b(s[i]);
}

// ---------------------------------------------------------------- bf16 GEMM
// C[M,N] = A[M,K] * B[K,N], A/B bf16 row-major, fp32 accumulate.
// MODE 0: store fp32 to Cout.  MODE 1: scatter bf16 into Q[bhT64], K[bhT64],
// V transposed [bh,hd,T].
template <int MODE>
__global__ __launch_bounds__(256) void gemm_bf16(const u16* __restrict__ A,
                                                 const u16* __restrict__ Bm,
                                                 int M, int N, int K,
                                                 float* __restrict__ Cout,
                                                 u16* __restrict__ qo,
                                                 u16* __restrict__ ko,
                                                 u16* __restrict__ vo) {
    __shared__ __align__(32) u16 As[128 * 32];   // [m][k]
    __shared__ __align__(32) u16 Bs[128 * 32];   // [n][k] (transposed)

    const int tid  = threadIdx.x;
    const int lane = tid & 31;
    const int wave = tid >> 5;
    const int l16  = lane & 15;
    const int hgrp = lane >> 4;      // 0 or 1 (lane half)
    const int klo  = hgrp * 8;
    const int wm   = wave & 3;       // 4 waves in M
    const int wn   = wave >> 2;      // 2 waves in N
    const int m0   = blockIdx.y * 128;
    const int n0   = blockIdx.x * 128;

    v8f acc[2][4];
#pragma unroll
    for (int mt = 0; mt < 2; ++mt)
#pragma unroll
        for (int nt = 0; nt < 4; ++nt)
            acc[mt][nt] = v8f{0.f, 0.f, 0.f, 0.f, 0.f, 0.f, 0.f, 0.f};

    const int arow = tid >> 1, ahalf = tid & 1;   // A tile: 2 thr/row, 16 elems
    const int bkk  = tid >> 3, bns   = tid & 7;   // B tile: 8 thr/k-row, 16 n

    // precompute this thread's A-tile LDS destinations (wave-relative offsets)
    const uint32_t as_off0 =
        (uint32_t)(uintptr_t)&As[arow * 32 + ahalf * 16];
    const uint32_t as_off1 =
        (uint32_t)(uintptr_t)&As[arow * 32 + ahalf * 16 + 8];

    for (int kb = 0; kb < K; kb += 32) {
        __syncthreads();
        // stage A tile 128x32 via CDNA5 async copy (memory -> LDS direct)
        {
            const u16* ga = A + (size_t)(m0 + arow) * K + kb + ahalf * 16;
            async_load_b128(as_off0, ga);
            async_load_b128(as_off1, ga + 8);
        }
        // stage B tile 32x128 -> transposed [n][k] (register transpose)
        {
            const u16* gb = Bm + (size_t)(kb + bkk) * N + n0 + bns * 16;
            u16 tmp[16];
            *(uint4*)&tmp[0] = *(const uint4*)gb;
            *(uint4*)&tmp[8] = *(const uint4*)(gb + 8);
#pragma unroll
            for (int i = 0; i < 16; ++i)
                Bs[(bns * 16 + i) * 32 + bkk] = tmp[i];
        }
        if (kb + 32 < K) {
            __builtin_prefetch(A + (size_t)(m0 + arow) * K + kb + 32 + ahalf * 16, 0, 1);
            __builtin_prefetch(Bm + (size_t)(kb + 32 + bkk) * N + n0 + bns * 16, 0, 1);
        }
        wait_asynccnt0();      // our async A transfers done before the barrier
        __syncthreads();

        ABfrag af[2];
#pragma unroll
        for (int mt = 0; mt < 2; ++mt) {
            const int base = (wm * 32 + mt * 16 + l16) * 32;
            af[mt].q[0] = *(const uint4*)&As[base + hgrp * 16];
            af[mt].q[1] = *(const uint4*)&As[base + hgrp * 16 + 8];
        }
        v16bf bf[4];
#pragma unroll
        for (int nt = 0; nt < 4; ++nt)
            bf[nt] = *(const v16bf*)&Bs[(wn * 64 + nt * 16 + l16) * 32 + hgrp * 16];

#pragma unroll
        for (int mt = 0; mt < 2; ++mt)
#pragma unroll
            for (int nt = 0; nt < 4; ++nt)
                acc[mt][nt] = wmma_bf16(af[mt].v, bf[nt], acc[mt][nt]);
    }

    // epilogue
#pragma unroll
    for (int mt = 0; mt < 2; ++mt) {
#pragma unroll
        for (int nt = 0; nt < 4; ++nt) {
#pragma unroll
            for (int j = 0; j < 8; ++j) {
                const int row = m0 + wm * 32 + mt * 16 + j + 8 * hgrp;
                const int col = n0 + wn * 64 + nt * 16 + l16;
                const float val = acc[mt][nt][j];
                if (MODE == 0) {
                    Cout[(size_t)row * N + col] = val;
                } else {
                    const int which = col >> 10;
                    const int d     = col & 1023;
                    const int h     = d >> 6;
                    const int hd    = d & 63;
                    const int bb    = row >> 11;       // /T
                    const int t     = row & 2047;      // %T
                    const size_t bh = (size_t)bb * Hn + h;
                    if (which == 0)
                        qo[(bh * Tn + t) * HDn + hd] = f2b(val);
                    else if (which == 1)
                        ko[(bh * Tn + t) * HDn + hd] = f2b(val);
                    else
                        vo[(bh * HDn + hd) * Tn + t] = f2b(val);
                }
            }
        }
    }
}

// ---------------------------------------------------------------- attention
// grid: (B*H, T/128). 8 waves/block, each wave owns 16 q-rows, full HD=64.
__global__ __launch_bounds__(256) void attn_kernel(const u16* __restrict__ qws,
                                                   const u16* __restrict__ kws,
                                                   const u16* __restrict__ vws,
                                                   const int* __restrict__ seg,
                                                   u16* __restrict__ yws) {
    __shared__ __align__(32) u16 Pl[8][16][32];

    const int bh   = blockIdx.x;
    const int b    = bh / Hn;
    const int h    = bh % Hn;
    const int tid  = threadIdx.x;
    const int lane = tid & 31;
    const int wave = tid >> 5;
    const int l16  = lane & 15;
    const int hgrp = lane >> 4;
    const int klo  = hgrp * 8;
    const int q0   = blockIdx.y * 128 + wave * 16;

    const u16* qbase = qws + (size_t)bh * Tn * HDn;
    const u16* kbase = kws + (size_t)bh * Tn * HDn;
    const u16* vbase = vws + (size_t)bh * HDn * Tn;
    const int* segb  = seg + (size_t)b * Tn;

    // Q A-frags: two K=32 chunks covering HD=64
    ABfrag aq[2];
#pragma unroll
    for (int c = 0; c < 2; ++c) {
        const size_t base = (size_t)(q0 + l16) * HDn + c * 32;
        aq[c].q[0] = *(const uint4*)&qbase[base + hgrp * 16];
        aq[c].q[1] = *(const uint4*)&qbase[base + hgrp * 16 + 8];
    }

    int qseg[8];
#pragma unroll
    for (int j = 0; j < 8; ++j) qseg[j] = segb[q0 + j + 8 * hgrp];

    float m[8], l[8];
#pragma unroll
    for (int j = 0; j < 8; ++j) { m[j] = NEGBIG; l[j] = 0.f; }
    v8f o0 = {0.f,0.f,0.f,0.f,0.f,0.f,0.f,0.f};
    v8f o1 = o0, o2 = o0, o3 = o0;

    const float sc = 0.125f * 1.44269504f;   // 1/sqrt(64) * log2(e)
    const int kbmax = (q0 + 16 + 31) >> 5;
    u16 (*pw)[32] = Pl[wave];

    for (int kb = 0; kb < kbmax; ++kb) {
        const int k0 = kb * 32;

        // S = Q * K^T : two 16x16 tiles (keys split 2x16), K-dim = HD in 2 chunks
        v8f s0 = {0.f,0.f,0.f,0.f,0.f,0.f,0.f,0.f};
        v8f s1 = s0;
#pragma unroll
        for (int c = 0; c < 2; ++c) {
            v16bf bk0 = *(const v16bf*)&kbase[(size_t)(k0 + l16) * HDn + c * 32 + hgrp * 16];
            v16bf bk1 = *(const v16bf*)&kbase[(size_t)(k0 + 16 + l16) * HDn + c * 32 + hgrp * 16];
            s0 = wmma_bf16(aq[c].v, bk0, s0);
            s1 = wmma_bf16(aq[c].v, bk1, s1);
        }

        const int kcol0 = k0 + l16;
        const int kcol1 = k0 + 16 + l16;
        const int ks0 = segb[kcol0];
        const int ks1 = segb[kcol1];

#pragma unroll
        for (int j = 0; j < 8; ++j) {
            const int   qr = q0 + j + 8 * hgrp;
            const int   qs = qseg[j];
            float v0 = s0[j] * sc;
            float v1 = s1[j] * sc;
            v0 = ((kcol0 <= qr) && (ks0 == qs)) ? v0 : NEGBIG;
            v1 = ((kcol1 <= qr) && (ks1 == qs)) ? v1 : NEGBIG;
            float rm = fmaxf(v0, v1);
            rm = fmaxf(rm, __shfl_xor(rm, 1, 32));
            rm = fmaxf(rm, __shfl_xor(rm, 2, 32));
            rm = fmaxf(rm, __shfl_xor(rm, 4, 32));
            rm = fmaxf(rm, __shfl_xor(rm, 8, 32));
            const float mn    = fmaxf(m[j], rm);
            const float alpha = exp2f(m[j] - mn);
            const float ms    = fmaxf(mn, -1.0e30f);
            m[j] = mn;
            const float e0 = exp2f(v0 - ms);
            const float e1 = exp2f(v1 - ms);
            float rs = e0 + e1;
            rs += __shfl_xor(rs, 1, 32);
            rs += __shfl_xor(rs, 2, 32);
            rs += __shfl_xor(rs, 4, 32);
            rs += __shfl_xor(rs, 8, 32);
            l[j] = l[j] * alpha + rs;
            o0[j] *= alpha; o1[j] *= alpha; o2[j] *= alpha; o3[j] *= alpha;
            pw[j + 8 * hgrp][l16]      = f2b(e0);   // C-layout -> LDS
            pw[j + 8 * hgrp][16 + l16] = f2b(e1);
        }

        // reload P as A-frag (wave-private LDS round-trip transposes layout)
        ABfrag ap;
        ap.q[0] = *(const uint4*)&pw[l16][hgrp * 16];
        ap.q[1] = *(const uint4*)&pw[l16][hgrp * 16 + 8];

        // O += P * V  (V transposed [hd][t]: contiguous keys per hd row)
        v16bf bv0 = *(const v16bf*)&vbase[(size_t)(0 * 16 + l16) * Tn + k0 + hgrp * 16];
        v16bf bv1 = *(const v16bf*)&vbase[(size_t)(1 * 16 + l16) * Tn + k0 + hgrp * 16];
        v16bf bv2 = *(const v16bf*)&vbase[(size_t)(2 * 16 + l16) * Tn + k0 + hgrp * 16];
        v16bf bv3 = *(const v16bf*)&vbase[(size_t)(3 * 16 + l16) * Tn + k0 + hgrp * 16];
        o0 = wmma_bf16(ap.v, bv0, o0);
        o1 = wmma_bf16(ap.v, bv1, o1);
        o2 = wmma_bf16(ap.v, bv2, o2);
        o3 = wmma_bf16(ap.v, bv3, o3);
    }

    // epilogue: normalize and store bf16 Y[b,t, h*64+hd]
#pragma unroll
    for (int j = 0; j < 8; ++j) {
        const float rl = 1.0f / l[j];
        const int   t  = q0 + j + 8 * hgrp;
        u16* yrow = yws + ((size_t)b * Tn + t) * Dn + h * HDn;
        yrow[0 * 16 + l16] = f2b(o0[j] * rl);
        yrow[1 * 16 + l16] = f2b(o1[j] * rl);
        yrow[2 * 16 + l16] = f2b(o2[j] * rl);
        yrow[3 * 16 + l16] = f2b(o3[j] * rl);
    }
}

// ---------------------------------------------------------------- launch
extern "C" void kernel_launch(void* const* d_in, const int* in_sizes, int n_in,
                              void* d_out, int out_size, void* d_ws, size_t ws_size,
                              hipStream_t stream) {
    (void)in_sizes; (void)n_in; (void)out_size; (void)ws_size;
    const float* x     = (const float*)d_in[0];
    const int*   seg   = (const int*)d_in[1];
    const float* wqkv  = (const float*)d_in[2];
    const float* wproj = (const float*)d_in[3];
    float*       out   = (float*)d_out;

    const size_t NX = (size_t)Bn * Tn * Dn;        // 4096*1024
    u16* xb  = (u16*)d_ws;
    u16* wqb = xb  + NX;
    u16* wpb = wqb + (size_t)Dn * 3 * Dn;
    u16* qw  = wpb + (size_t)Dn * Dn;
    u16* kw  = qw  + NX;
    u16* vw  = kw  + NX;
    u16* yb  = vw  + NX;

    // fp32 -> bf16 conversions
    cvt_f32_bf16<<<dim3((unsigned)((NX + 255) / 256)), 256, 0, stream>>>(x, xb, (int)NX);
    cvt_f32_bf16<<<dim3((unsigned)((Dn * 3 * Dn + 255) / 256)), 256, 0, stream>>>(
        wqkv, wqb, Dn * 3 * Dn);
    cvt_f32_bf16<<<dim3((unsigned)((Dn * Dn + 255) / 256)), 256, 0, stream>>>(
        wproj, wpb, Dn * Dn);

    // QKV projection: [4096,1024] x [1024,3072] -> scatter Q/K/V^T bf16
    gemm_bf16<1><<<dim3(3 * Dn / 128, (unsigned)(NX / Dn / 128)), 256, 0, stream>>>(
        xb, wqb, (int)(NX / Dn), 3 * Dn, Dn, nullptr, qw, kw, vw);

    // flash attention per (b,h), q-blocks of 128
    attn_kernel<<<dim3(Bn * Hn, Tn / 128), 256, 0, stream>>>(qw, kw, vw, seg, yb);

    // output projection: [4096,1024] x [1024,1024] -> fp32 out
    gemm_bf16<0><<<dim3(Dn / 128, (unsigned)(NX / Dn / 128)), 256, 0, stream>>>(
        yb, wpb, (int)(NX / Dn), Dn, Dn, out, nullptr, nullptr, nullptr);
}
